// BertSelfAttention_30820685316250
// MI455X (gfx1250) — matrix operand order repaired and checked by reading, not compiled, not run on hardware
//
#include <hip/hip_runtime.h>
#include <stdint.h>

#define HIDDEN   1024
#define HEADS    16
#define HEAD_DIM 64
#define RANK     16
#define SEQ      2048
#define BS       4
#define NROWS    (BS * SEQ)          // 8192
#define LORA_SCALE (1.0f / 16.0f)

typedef __bf16 v8bf  __attribute__((ext_vector_type(8)));
typedef __bf16 v16bf __attribute__((ext_vector_type(16)));
typedef float  v8f   __attribute__((ext_vector_type(8)));
typedef int    v4i_t __attribute__((ext_vector_type(4)));

union AFrag { v16bf v; v8bf h[2]; };
union CFrag { v8f v; float f[8]; };

// ---- gfx1250 async global->LDS copy (ASYNCcnt-tracked), guarded ----
#if defined(__AMDGCN__) && __has_builtin(__builtin_amdgcn_global_load_async_to_lds_b128)
#define HAVE_ASYNC_LDS 1
#else
#define HAVE_ASYNC_LDS 0
#endif

#if HAVE_ASYNC_LDS
// Prototype (from clang diagnostic): (int4* gsrc, int4* ldst, imm offset, imm cpol)
#define ASYNC_CP_B128(gp, lp)                                                  \
  __builtin_amdgcn_global_load_async_to_lds_b128((v4i_t*)(gp), (v4i_t*)(lp), 0, 0)
#if __has_builtin(__builtin_amdgcn_s_wait_asynccnt)
#define WAIT_ASYNC(n) __builtin_amdgcn_s_wait_asynccnt(n)
#else
#define WAIT_ASYNC(n) asm volatile("s_wait_asynccnt %0" ::"i"(n) : "memory")
#endif
#else
#define WAIT_ASYNC(n)
#endif

__device__ __forceinline__ uint16_t f2bf(float f) {
  uint32_t u = __float_as_uint(f);
  u += 0x7FFFu + ((u >> 16) & 1u);   // round-to-nearest-even
  return (uint16_t)(u >> 16);
}

// ---------------------------------------------------------------------------
// Kernel 1: fp32 -> bf16 convert of hidden_states, row-major [8192][1024]
// ---------------------------------------------------------------------------
__global__ void convert_x_bf16(const float* __restrict__ x, uint16_t* __restrict__ xb) {
  int i = blockIdx.x * blockDim.x + threadIdx.x;
  xb[i] = f2bf(x[i]);
}

// ---------------------------------------------------------------------------
// Kernel 2: fold LoRA into weights: Weff[z][o][i] = W[o][i] + (1/16)*(B A)[o][i]
// ---------------------------------------------------------------------------
__global__ void prep_weights(const float* __restrict__ Wq, const float* __restrict__ Aq, const float* __restrict__ Bq,
                             const float* __restrict__ Wk, const float* __restrict__ Ak, const float* __restrict__ Bk,
                             const float* __restrict__ Wv, const float* __restrict__ Av, const float* __restrict__ Bv,
                             uint16_t* __restrict__ weff) {
  int t   = blockIdx.x * blockDim.x + threadIdx.x;
  int z   = t >> 20;
  int rem = t & ((1 << 20) - 1);
  int o   = rem >> 10;
  int i   = rem & 1023;
  const float* W = (z == 0) ? Wq : (z == 1) ? Wk : Wv;
  const float* A = (z == 0) ? Aq : (z == 1) ? Ak : Av;
  const float* B = (z == 0) ? Bq : (z == 1) ? Bk : Bv;
  float acc = 0.f;
#pragma unroll
  for (int r = 0; r < RANK; ++r) acc += B[o * RANK + r] * A[r * HIDDEN + i];
  weff[t] = f2bf(W[o * HIDDEN + i] + LORA_SCALE * acc);
}

// ---------------------------------------------------------------------------
// Kernel 3: QKV projection GEMM, bf16 WMMA 16x16x32, f32 accum.
// ---------------------------------------------------------------------------
__global__ __launch_bounds__(256) void qkv_proj(
    const uint16_t* __restrict__ xb, const uint16_t* __restrict__ weff,
    const float* __restrict__ bq, const float* __restrict__ bk, const float* __restrict__ bv,
    uint16_t* __restrict__ Qb, uint16_t* __restrict__ Kb, uint16_t* __restrict__ Vt) {
  const int lane  = threadIdx.x & 31;
  const int wid   = threadIdx.x >> 5;
  const int ln    = lane & 15;
  const int hl    = lane >> 4;
  const int z     = blockIdx.z;
  const int mrow0 = blockIdx.x * 16;
  const int ncol0 = blockIdx.y * 512 + wid * 64;

  const uint16_t* Wz   = weff + (size_t)z * HIDDEN * HIDDEN;
  const float*    bias = (z == 0) ? bq : (z == 1) ? bk : bv;

  CFrag c[4];
#pragma unroll
  for (int j = 0; j < 4; ++j) {
    float bb = bias[ncol0 + j * 16 + ln];
#pragma unroll
    for (int r = 0; r < 8; ++r) c[j].f[r] = bb;
  }

  const uint16_t* xrow = xb + (size_t)(mrow0 + ln) * HIDDEN;
  for (int k0 = 0; k0 < HIDDEN; k0 += 32) {
    AFrag a;
    a.h[0] = *(const v8bf*)(xrow + k0 + hl * 16);
    a.h[1] = *(const v8bf*)(xrow + k0 + hl * 16 + 8);
#pragma unroll
    for (int j = 0; j < 4; ++j) {
      v16bf bfrag = *(const v16bf*)(Wz + (size_t)(ncol0 + j * 16 + ln) * HIDDEN + k0 + hl * 16);
      c[j].v = __builtin_amdgcn_wmma_f32_16x16x32_bf16(false, a.v, false, bfrag,
                                                       (short)0, c[j].v, false, false);
    }
  }

#pragma unroll
  for (int j = 0; j < 4; ++j) {
    int ocol = ncol0 + j * 16 + ln;
    int hh   = ocol >> 6;
    int dd   = ocol & 63;
#pragma unroll
    for (int r = 0; r < 8; ++r) {
      int g = mrow0 + r + 8 * hl;       // global row
      int b = g >> 11;                  // batch
      int s = g & 2047;                 // seq pos
      uint16_t val = f2bf(c[j].f[r]);
      if (z == 0)      Qb[(((size_t)(b * HEADS + hh)) * SEQ + s) * HEAD_DIM + dd] = val;
      else if (z == 1) Kb[(((size_t)(b * HEADS + hh)) * SEQ + s) * HEAD_DIM + dd] = val;
      else             Vt[(((size_t)(b * HEADS + hh)) * HEAD_DIM + dd) * SEQ + s] = val;
    }
  }
}

// ---------------------------------------------------------------------------
// Kernel 4: flash attention with WG-shared, double-buffered async LDS staging
// of K/V tiles (global_load_async_to_lds_b128, ASYNCcnt), prefetching tile
// k+1 during tile k's WMMA + softmax. Per-wave: S = QK^T (4 wmma from LDS
// fragments), online softmax (shfl_xor reductions), P -> LDS -> A-layout,
// P*V (4 wmma).
// ---------------------------------------------------------------------------
__global__ __launch_bounds__(256) void flash_attn(
    const uint16_t* __restrict__ Qb, const uint16_t* __restrict__ Kb,
    const uint16_t* __restrict__ Vt, const float* __restrict__ mask,
    float* __restrict__ out) {
  __shared__ uint16_t Ktile[2][32 * 64];   // [buf][key][dim]
  __shared__ uint16_t Vtile[2][64 * 32];   // [buf][dim][key]
  __shared__ uint16_t Pbuf[8][16 * 32];    // per-wave P transpose buffer

  const int tid   = threadIdx.x;
  const int lane  = tid & 31;
  const int wid   = tid >> 5;
  const int ln    = lane & 15;
  const int hl    = lane >> 4;
  const int bh    = blockIdx.x;
  const int b     = bh >> 4;
  const int head  = bh & 15;
  const int qrow0 = blockIdx.y * 128 + wid * 16;

  uint16_t* pbuf = &Pbuf[wid][0];
  const uint16_t* Kbase = Kb + (size_t)bh * SEQ * HEAD_DIM;
  const uint16_t* Vbase = Vt + (size_t)bh * HEAD_DIM * SEQ;

  // Stage one 32-key tile (K: 4KB, V: 4KB) into LDS; 2 x b128 per thread.
  auto stage = [&](int buf, int n0) {
    const uint16_t* gk = Kbase + (size_t)(n0 + (tid >> 3)) * HEAD_DIM + (tid & 7) * 8;
    uint16_t*       lk = &Ktile[buf][(tid >> 3) * 64 + (tid & 7) * 8];
    const uint16_t* gv = Vbase + (size_t)(tid >> 2) * SEQ + n0 + (tid & 3) * 8;
    uint16_t*       lv = &Vtile[buf][(tid >> 2) * 32 + (tid & 3) * 8];
#if HAVE_ASYNC_LDS
    ASYNC_CP_B128(gk, lk);
    ASYNC_CP_B128(gv, lv);
#else
    *(v8bf*)lk = *(const v8bf*)gk;
    *(v8bf*)lv = *(const v8bf*)gv;
#endif
  };

  // Q A-fragments for head_dim chunks [0,32) and [32,64)
  const uint16_t* qrow = Qb + (((size_t)bh * SEQ) + qrow0 + ln) * HEAD_DIM;
  AFrag aq0, aq1;
  aq0.h[0] = *(const v8bf*)(qrow + hl * 16);
  aq0.h[1] = *(const v8bf*)(qrow + hl * 16 + 8);
  aq1.h[0] = *(const v8bf*)(qrow + 32 + hl * 16);
  aq1.h[1] = *(const v8bf*)(qrow + 32 + hl * 16 + 8);

  CFrag o[4];
  float rm[8], rl[8];
#pragma unroll
  for (int j = 0; j < 4; ++j)
#pragma unroll
    for (int r = 0; r < 8; ++r) o[j].f[r] = 0.f;
#pragma unroll
  for (int r = 0; r < 8; ++r) { rm[r] = -3.0e38f; rl[r] = 0.f; }

  const float* mrow = mask + (size_t)b * SEQ;

  stage(0, 0);
  for (int kt = 0; kt < SEQ / 32; ++kt) {
    const int buf = kt & 1;
    const int n0  = kt * 32;
    if (kt + 1 < SEQ / 32) {
      stage(buf ^ 1, n0 + 32);   // prefetch next tile into other buffer
      WAIT_ASYNC(2);             // retire current tile's 2 copies; keep prefetch in flight
    } else {
      WAIT_ASYNC(0);
    }
    __syncthreads();             // all waves' current-tile data visible in LDS

    // ---- S = Q K^T for 2 tiles of 16 keys (fragments from LDS) ----
    const uint16_t* kl = &Ktile[buf][0];
    const uint16_t* vl = &Vtile[buf][0];
    CFrag s0, s1;
    {
      v16bf b00 = *(const v16bf*)(kl + ln * 64 + hl * 16);
      v16bf b01 = *(const v16bf*)(kl + ln * 64 + 32 + hl * 16);
      v16bf b10 = *(const v16bf*)(kl + (16 + ln) * 64 + hl * 16);
      v16bf b11 = *(const v16bf*)(kl + (16 + ln) * 64 + 32 + hl * 16);
      v8f zero = {};
      s0.v = __builtin_amdgcn_wmma_f32_16x16x32_bf16(false, aq0.v, false, b00, (short)0, zero, false, false);
      s0.v = __builtin_amdgcn_wmma_f32_16x16x32_bf16(false, aq1.v, false, b01, (short)0, s0.v, false, false);
      s1.v = __builtin_amdgcn_wmma_f32_16x16x32_bf16(false, aq0.v, false, b10, (short)0, zero, false, false);
      s1.v = __builtin_amdgcn_wmma_f32_16x16x32_bf16(false, aq1.v, false, b11, (short)0, s1.v, false, false);
    }
    float m0 = mrow[n0 + ln];
    float m1 = mrow[n0 + 16 + ln];
#pragma unroll
    for (int r = 0; r < 8; ++r) {
      s0.f[r] = s0.f[r] * 0.125f + m0;
      s1.f[r] = s1.f[r] * 0.125f + m1;
    }

    // ---- row max over 32 keys ----
    float tm[8];
#pragma unroll
    for (int r = 0; r < 8; ++r) tm[r] = fmaxf(s0.f[r], s1.f[r]);
#pragma unroll
    for (int off = 1; off < 16; off <<= 1)
#pragma unroll
      for (int r = 0; r < 8; ++r) tm[r] = fmaxf(tm[r], __shfl_xor(tm[r], off, 32));

    float p0[8], p1[8], ts[8], corr[8];
#pragma unroll
    for (int r = 0; r < 8; ++r) {
      float mn = fmaxf(rm[r], tm[r]);
      corr[r]  = __expf(rm[r] - mn);
      rm[r]    = mn;
      p0[r]    = __expf(s0.f[r] - mn);
      p1[r]    = __expf(s1.f[r] - mn);
      ts[r]    = p0[r] + p1[r];
      rl[r]   *= corr[r];
    }
#pragma unroll
    for (int off = 1; off < 16; off <<= 1)
#pragma unroll
      for (int r = 0; r < 8; ++r) ts[r] += __shfl_xor(ts[r], off, 32);
#pragma unroll
    for (int r = 0; r < 8; ++r) rl[r] += ts[r];
#pragma unroll
    for (int j = 0; j < 4; ++j)
#pragma unroll
      for (int r = 0; r < 8; ++r) o[j].f[r] *= corr[r];

    // ---- P (D-layout) -> LDS bf16 -> reload in A-layout ----
    asm volatile("s_wait_dscnt 0x0" ::: "memory");
#pragma unroll
    for (int r = 0; r < 8; ++r) {
      pbuf[(r + 8 * hl) * 32 + ln]      = f2bf(p0[r]);
      pbuf[(r + 8 * hl) * 32 + 16 + ln] = f2bf(p1[r]);
    }
    asm volatile("s_wait_dscnt 0x0" ::: "memory");
    AFrag ap;
    ap.h[0] = *(const v8bf*)(pbuf + ln * 32 + hl * 16);
    ap.h[1] = *(const v8bf*)(pbuf + ln * 32 + hl * 16 + 8);

    // ---- O += P * V  (V fragments from LDS, key-contiguous) ----
#pragma unroll
    for (int j = 0; j < 4; ++j) {
      v16bf bvf = *(const v16bf*)(vl + (j * 16 + ln) * 32 + hl * 16);
      o[j].v = __builtin_amdgcn_wmma_f32_16x16x32_bf16(false, ap.v, false, bvf, (short)0, o[j].v, false, false);
    }

    __syncthreads();             // everyone done reading buf before it's overwritten
  }

  // ---- normalize and store fp32 output [b][seq][h*64+d] ----
  float inv[8];
#pragma unroll
  for (int r = 0; r < 8; ++r) inv[r] = 1.0f / rl[r];
#pragma unroll
  for (int j = 0; j < 4; ++j)
#pragma unroll
    for (int r = 0; r < 8; ++r) {
      int row = qrow0 + r + 8 * hl;
      int col = head * 64 + j * 16 + ln;
      out[((size_t)b * SEQ + row) * HIDDEN + col] = o[j].f[r] * inv[r];
    }
}

// ---------------------------------------------------------------------------
extern "C" void kernel_launch(void* const* d_in, const int* in_sizes, int n_in,
                              void* d_out, int out_size, void* d_ws, size_t ws_size,
                              hipStream_t stream) {
  (void)in_sizes; (void)n_in; (void)out_size; (void)ws_size;
  const float* hs   = (const float*)d_in[0];
  const float* mask = (const float*)d_in[1];
  const float* Wq = (const float*)d_in[2],  *bq = (const float*)d_in[3];
  const float* Aq = (const float*)d_in[4],  *Bq = (const float*)d_in[5];
  const float* Wk = (const float*)d_in[6],  *bk = (const float*)d_in[7];
  const float* Ak = (const float*)d_in[8],  *Bk = (const float*)d_in[9];
  const float* Wv = (const float*)d_in[10], *bv = (const float*)d_in[11];
  const float* Av = (const float*)d_in[12], *Bv = (const float*)d_in[13];

  uint16_t* Xb   = (uint16_t*)d_ws;                       // 8192*1024
  uint16_t* Weff = Xb   + (size_t)NROWS * HIDDEN;         // 3*1024*1024
  uint16_t* Qb   = Weff + (size_t)3 * HIDDEN * HIDDEN;    // 8192*1024
  uint16_t* Kb   = Qb   + (size_t)NROWS * HIDDEN;
  uint16_t* Vt   = Kb   + (size_t)NROWS * HIDDEN;

  convert_x_bf16<<<(NROWS * HIDDEN) / 256, 256, 0, stream>>>(hs, Xb);
  prep_weights<<<(3 * HIDDEN * HIDDEN) / 256, 256, 0, stream>>>(
      Wq, Aq, Bq, Wk, Ak, Bk, Wv, Av, Bv, Weff);
  qkv_proj<<<dim3(NROWS / 16, 2, 3), 256, 0, stream>>>(
      Xb, Weff, bq, bk, bv, Qb, Kb, Vt);
  flash_attn<<<dim3(BS * HEADS, SEQ / 128), 256, 0, stream>>>(
      Qb, Kb, Vt, mask, (float*)d_out);
}
